// ThermoGRL_82686710383108
// MI455X (gfx1250) — matrix operand order repaired and checked by reading, not compiled, not run on hardware
//
#include <hip/hip_runtime.h>
#include <stdint.h>

// ---------------------------------------------------------------------------
// CDNA5 / gfx1250 GNN forward:
//   h  = relu(x @ W_enc + b_enc)                     (WMMA bf16, f32 A fused)
//   xw = h @ W_gcn                                   (WMMA bf16 GEMM)
//   agg = scatter_add(xw[src]*dinv[src]*dinv[dst]) + xw*dinv^2
//   h2 = relu(agg + b_gcn)
//   q1 = relu(h2 @ W1 + b1)                          (WMMA bf16 GEMM)
//   q  = q1 @ W2 + b2                                (WMMA bf16 GEMM, N=10)
// ---------------------------------------------------------------------------

typedef __attribute__((ext_vector_type(16))) __bf16 v16bf;
typedef __attribute__((ext_vector_type(8)))  float  v8f;

union Frag16 {
  v16bf v;
  uint4 q[2];
  __bf16 e[16];
};

static __device__ __forceinline__ float bf2f(unsigned short h) {
  return __uint_as_float(((unsigned)h) << 16);
}
static __device__ __forceinline__ unsigned short f2bf(float x) {
  unsigned u = __float_as_uint(x);
  unsigned r = u + 0x7FFFu + ((u >> 16) & 1u);  // RNE
  return (unsigned short)(r >> 16);
}
static __device__ __forceinline__ void cvt8(__bf16* d, float4 a, float4 b) {
  d[0] = (__bf16)a.x; d[1] = (__bf16)a.y; d[2] = (__bf16)a.z; d[3] = (__bf16)a.w;
  d[4] = (__bf16)b.x; d[5] = (__bf16)b.y; d[6] = (__bf16)b.z; d[7] = (__bf16)b.w;
}

// ---------------------------------------------------------------------------
// WMMA GEMM:  out[M, LDOUT] = act(A[M, KDIM] * W[KDIM, NW](f32) + bias)
// A is bf16, or f32 converted in-register when AF32 (fuses the x->bf16 pass).
// 4 waves / block, each wave owns a 16-row strip across NT 16-col tiles
// starting at column nOffset. W is staged in LDS pre-swizzled into the exact
// B-fragment layout (lane = n%16 + 16*((k%16)>=8), 16 contiguous bf16 / lane).
// NW/LDOUT/NSTORE compile-time -> store offsets are instruction immediates.
// ---------------------------------------------------------------------------
template <int KDIM, int NT, int NW, int LDOUT, int NSTORE, bool RELU,
          bool HASBIAS, bool OUTBF16, bool AF32>
__global__ __launch_bounds__(128) void wmma_gemm(
    const void* __restrict__ A,   // [M, KDIM] bf16 (or f32 if AF32)
    const float* __restrict__ W,  // [KDIM, NW] f32 row-major
    const float* __restrict__ bias,
    void* __restrict__ out,       // bf16 or f32, [M, LDOUT]
    int M, int nOffset) {
  extern __shared__ uint4 smem4[];
  __bf16* wlds = (__bf16*)smem4;

  const int tid = threadIdx.x;
  const int wid = tid >> 5;
  const int lane = tid & 31;
  const int lmod = lane & 15;
  const int lhalf = lane >> 4;
  const int m0 = (blockIdx.x * 4 + wid) * 16;

  int rowA = m0 + lmod;
  if (rowA >= M) rowA = M - 1;

  // A fragments for the whole K dimension, held in VGPRs.
  Frag16 a[KDIM / 32];
  if (AF32) {
    const float* ar = (const float*)A + (size_t)rowA * KDIM + 8 * lhalf;
#pragma unroll
    for (int kt = 0; kt < KDIM / 32; ++kt)
      __builtin_prefetch(ar + kt * 32, 0, 3);  // global_prefetch_b8
#pragma unroll
    for (int kt = 0; kt < KDIM / 32; ++kt) {
      float4 f0 = *(const float4*)(ar + kt * 32);
      float4 f1 = *(const float4*)(ar + kt * 32 + 4);
      float4 f2 = *(const float4*)(ar + kt * 32 + 16);
      float4 f3 = *(const float4*)(ar + kt * 32 + 20);
      cvt8(&a[kt].e[0], f0, f1);   // elems 0..7  : K = k0 + 8*lhalf + j
      cvt8(&a[kt].e[8], f2, f3);   // elems 8..15 : K = k0 + 16 + 8*lhalf + j
    }
  } else {
    const unsigned short* ar =
        (const unsigned short*)A + (size_t)rowA * KDIM + 8 * lhalf;
#pragma unroll
    for (int kt = 0; kt < KDIM / 32; ++kt)
      __builtin_prefetch(ar + kt * 32, 0, 3);
#pragma unroll
    for (int kt = 0; kt < KDIM / 32; ++kt) {
      a[kt].q[0] = *(const uint4*)(ar + kt * 32);
      a[kt].q[1] = *(const uint4*)(ar + kt * 32 + 16);
    }
  }

  // Stage W into LDS in B-fragment order (bf16).
  const int TOT = KDIM * NT * 16;
  for (int f = tid; f < TOT; f += blockDim.x) {
    int j = f & 15;          // element within lane (k = 8*half + j%8 + 16*(j/8))
    int ls = (f >> 4) & 31;  // lane slot
    int fi = f >> 9;         // fragment index = kt*NT + nt
    int kt = fi / NT;
    int nt = fi - kt * NT;
    int k = kt * 32 + (ls >> 4) * 8 + (j & 7) + (j >> 3) * 16;
    int n = nOffset + nt * 16 + (ls & 15);
    float w = (n < NW) ? W[(size_t)k * NW + n] : 0.0f;
    wlds[f] = (__bf16)w;
  }
  __syncthreads();

#pragma unroll
  for (int nt = 0; nt < NT; ++nt) {
    const int col = nOffset + nt * 16 + lmod;  // this lane's output column
    float bv = 0.0f;
    if (HASBIAS) {
      if ((NW & 15) == 0 || col < NW) bv = bias[col];
    }
    v8f acc = {};
#pragma unroll
    for (int r = 0; r < 8; ++r) acc[r] = bv;  // bias folded into C

#pragma unroll
    for (int kt = 0; kt < KDIM / 32; ++kt) {
      Frag16 b;
      const uint4* bp = smem4 + ((size_t)((kt * NT + nt) * 32 + lane)) * 2;
      b.q[0] = bp[0];
      b.q[1] = bp[1];
      acc = __builtin_amdgcn_wmma_f32_16x16x32_bf16(
          false, a[kt].v, false, b.v, (short)0, acc, false, false);
    }

    const int rowBase = m0 + lhalf * 8;
    const bool colOK = ((NSTORE & 15) == 0) || (col < NSTORE);
    if (colOK && (rowBase + 8 <= M)) {
      // Fast path: branchless, native bf16 converts, immediate store offsets.
      if (OUTBF16) {
        __bf16* p = (__bf16*)out + rowBase * LDOUT + col;
#pragma unroll
        for (int r = 0; r < 8; ++r) {
          float v = acc[r];
          if (RELU) v = fmaxf(v, 0.0f);
          p[r * LDOUT] = (__bf16)v;
        }
      } else {
        float* p = (float*)out + rowBase * LDOUT + col;
#pragma unroll
        for (int r = 0; r < 8; ++r) {
          float v = acc[r];
          if (RELU) v = fmaxf(v, 0.0f);
          p[r * LDOUT] = v;
        }
      }
    } else if (colOK) {
      // Boundary rows only (last block).
#pragma unroll
      for (int r = 0; r < 8; ++r) {
        int row = rowBase + r;
        if (row < M) {
          float v = acc[r];
          if (RELU) v = fmaxf(v, 0.0f);
          if (OUTBF16)
            ((__bf16*)out)[row * LDOUT + col] = (__bf16)v;
          else
            ((float*)out)[row * LDOUT + col] = v;
        }
      }
    }
  }
}

// ---------------------------------------------------------------------------
// Elementwise / edge kernels
// ---------------------------------------------------------------------------
__global__ void k_fill1(float* __restrict__ p, int n) {
  int i = blockIdx.x * blockDim.x + threadIdx.x;
  if (i < n) p[i] = 1.0f;
}

__global__ void k_deg_acc(const long long* __restrict__ ei,
                          float* __restrict__ deg, int E) {
  int e = blockIdx.x * blockDim.x + threadIdx.x;
  if (e < E) atomicAdd(&deg[(int)ei[(size_t)E + e]], 1.0f);
}

__global__ void k_rsqrt(const float* __restrict__ deg,
                        float* __restrict__ dinv, int n) {
  int i = blockIdx.x * blockDim.x + threadIdx.x;
  if (i < n) dinv[i] = rsqrtf(deg[i]);
}

// agg[i,f] = xw[i,f] * dinv[i]^2   (self-loop term, also initializes agg)
__global__ void k_selfagg(const unsigned short* __restrict__ xw,
                          const float* __restrict__ dinv,
                          float* __restrict__ agg, int n) {
  int i = blockIdx.x * blockDim.x + threadIdx.x;
  if (i < n) {
    float d = dinv[i >> 7];
    agg[i] = bf2f(xw[i]) * d * d;
  }
}

// One wave per edge: single b64 gather per lane (256B/wave), 4 contiguous
// f32 atomic adds (xw bf16 + agg f32 both stay resident in the 192MB L2).
__global__ void k_scatter(const unsigned short* __restrict__ xw,
                          const long long* __restrict__ ei,
                          const float* __restrict__ dinv,
                          float* __restrict__ agg, int E) {
  int lane = threadIdx.x & 31;
  int e = blockIdx.x * (blockDim.x >> 5) + (threadIdx.x >> 5);
  if (e >= E) return;
  int src = (int)ei[e];
  int dst = (int)ei[(size_t)E + e];
  float coef = dinv[src] * dinv[dst];
  union {
    uint2 u2;
    unsigned short s[4];
  } g;
  g.u2 = *(const uint2*)(xw + (size_t)src * 128 + lane * 4);
  float* ad = agg + (size_t)dst * 128 + lane * 4;
#pragma unroll
  for (int j = 0; j < 4; ++j) atomicAdd(&ad[j], bf2f(g.s[j]) * coef);
}

__global__ void k_bias_relu_bf16(const float* __restrict__ agg,
                                 const float* __restrict__ b,
                                 unsigned short* __restrict__ out, int n) {
  int i = blockIdx.x * blockDim.x + threadIdx.x;
  if (i < n) out[i] = f2bf(fmaxf(agg[i] + b[i & 127], 0.0f));
}

// ---------------------------------------------------------------------------
extern "C" void kernel_launch(void* const* d_in, const int* in_sizes, int n_in,
                              void* d_out, int out_size, void* d_ws,
                              size_t ws_size, hipStream_t stream) {
  const float*     x     = (const float*)d_in[0];
  const long long* ei    = (const long long*)d_in[1];  // int64 per reference
  const float*     W_enc = (const float*)d_in[2];
  const float*     b_enc = (const float*)d_in[3];
  const float*     W_gcn = (const float*)d_in[4];
  const float*     b_gcn = (const float*)d_in[5];
  const float*     W1    = (const float*)d_in[6];
  const float*     b1    = (const float*)d_in[7];
  const float*     W2    = (const float*)d_in[8];
  const float*     b2    = (const float*)d_in[9];
  float* q = (float*)d_out;

  const int N = in_sizes[0] / 128;  // 100000
  const int E = in_sizes[1] / 2;    // 3200000
  const int NF = N * 128;

  // Workspace layout (bf16 stored as ushort):
  //   [0]        h   : encoder output bf16   (reused later as h2)
  //   [1*szNF]   xw  : h @ W_gcn bf16
  //   [2*szNF]   agg : f32 N*128             (reused later as q1 bf16 N*256)
  //   [..]       deg, dinv : f32 N each
  char* ws = (char*)d_ws;
  const size_t szNF = (size_t)NF * 2;   // 25.6 MB
  const size_t szAgg = (size_t)NF * 4;  // 51.2 MB (== q1 bf16 N*256)
  unsigned short* h   = (unsigned short*)(ws);
  unsigned short* xw  = (unsigned short*)(ws + szNF);
  float*          agg = (float*)(ws + 2 * szNF);
  float*          deg = (float*)(ws + 2 * szNF + szAgg);
  float*          dinv = deg + N;
  unsigned short* h2 = h;                     // reuse (h dead after GEMM2)
  unsigned short* q1 = (unsigned short*)agg;  // reuse (agg dead after h2)

  const int TB = 256;
  const int gNF = (NF + TB - 1) / TB;
  const int gN = (N + TB - 1) / TB;
  const int gE = (E + TB - 1) / TB;
  const int gGemm = (N + 63) / 64;  // 4 waves * 16 rows per block
  const size_t shm128x128 = 128 * 128 * sizeof(unsigned short);  // 32 KB
  const size_t shm256x16  = 256 * 16 * sizeof(unsigned short);   // 8 KB

  // degrees: deg = 1 + in-degree ; dinv = rsqrt(deg)
  k_fill1<<<gN, TB, 0, stream>>>(deg, N);
  k_deg_acc<<<gE, TB, 0, stream>>>(ei, deg, E);
  k_rsqrt<<<gN, TB, 0, stream>>>(deg, dinv, N);

  // 1) h = relu(x @ W_enc + b_enc)   (f32 A converted in-register)
  wmma_gemm<128, 8, 128, 128, 128, true, true, true, true>
      <<<gGemm, 128, shm128x128, stream>>>(x, W_enc, b_enc, h, N, 0);

  // 2) xw = h @ W_gcn  (bias deferred until after aggregation)
  wmma_gemm<128, 8, 128, 128, 128, false, false, true, false>
      <<<gGemm, 128, shm128x128, stream>>>(h, W_gcn, nullptr, xw, N, 0);

  // 3) agg = xw * dinv^2  (self-loop, initializes agg)
  k_selfagg<<<gNF, TB, 0, stream>>>(xw, dinv, agg, NF);

  // 4) agg += scatter-add over edges
  k_scatter<<<(E + 7) / 8, 256, 0, stream>>>(xw, ei, dinv, agg, E);

  // 5) h2 = relu(agg + b_gcn)
  k_bias_relu_bf16<<<gNF, TB, 0, stream>>>(agg, b_gcn, h2, NF);

  // 6) q1 = relu(h2 @ W1 + b1)   (two 128-col halves -> 32 KB LDS each)
  wmma_gemm<128, 8, 256, 256, 256, true, true, true, false>
      <<<gGemm, 128, shm128x128, stream>>>(h2, W1, b1, q1, N, 0);
  wmma_gemm<128, 8, 256, 256, 256, true, true, true, false>
      <<<gGemm, 128, shm128x128, stream>>>(h2, W1, b1, q1, N, 128);

  // 7) q = q1 @ W2 + b2  (N padded 10->16 in LDS, store 10 cols, f32 out)
  wmma_gemm<256, 1, 10, 10, 10, false, true, false, false>
      <<<gGemm, 128, shm256x16, stream>>>(q1, W2, b2, q, N, 0);
}
